// Encoder_15788299780126
// MI455X (gfx1250) — compile-verified, matrix operand order checked by reading, not run on hardware
//
#include <hip/hip_runtime.h>
#include <hip/hip_bf16.h>

typedef __attribute__((ext_vector_type(16))) _Float16 v16h;
typedef __attribute__((ext_vector_type(8)))  float    v8f;
typedef __attribute__((ext_vector_type(4)))  unsigned int v4u;

union Frag { v4u q[2]; v16h h; };

// Problem constants (from reference setup_inputs)
constexpr int BSZ = 2048, NN = 64, OBS = 8, PRED = 12, INs = 6;
constexpr int EE = 512, CC = 6, CN = 7, KK = 20;
constexpr int DSELF = INs * (OBS + PRED);   // 120
constexpr int DSELF_P = 128;                // padded K for self GEMM
constexpr int DNEI = INs * OBS;             // 48
constexpr int DNEI_P = 64;                  // padded K for nei GEMM
constexpr int WST_ELEMS = CC * EE * DSELF_P;   // 393216 f16
constexpr int WNT_ELEMS = CN * EE * DNEI_P;    // 229376 f16
constexpr size_t DUMP_OFF = (size_t)(WST_ELEMS + WNT_ELEMS) * 2; // bytes into ws

// ---------------------------------------------------------------------------
// Kernel 0: convert fp32 weights -> f16, transposed [class][E][Kpad] so that a
// WMMA B-fragment (fixed N column, contiguous K) is two 16B global loads/lane.
// Weights total ~1.2MB f16 -> stay resident in the 192MB L2 for all 2048 WGs.
// ---------------------------------------------------------------------------
__global__ __launch_bounds__(256) void prep_weights_kernel(
    const float* __restrict__ Wself, const float* __restrict__ Wnei,
    _Float16* __restrict__ Wst, _Float16* __restrict__ Wnt)
{
    int i = blockIdx.x * 256 + threadIdx.x;
    if (i < WST_ELEMS) {
        int c = i / (EE * DSELF_P);
        int r = i % (EE * DSELF_P);
        int e = r >> 7;          // column (E)
        int k = r & 127;         // padded K
        float v = (k < DSELF) ? Wself[((size_t)c * DSELF + k) * EE + e] : 0.0f;
        Wst[i] = (_Float16)v;
    } else if (i < WST_ELEMS + WNT_ELEMS) {
        int j = i - WST_ELEMS;
        int c = j / (EE * DNEI_P);
        int r = j % (EE * DNEI_P);
        int e = r >> 6;
        int k = r & 63;
        float v = (k < DNEI) ? Wnei[((size_t)c * DNEI + k) * EE + e] : 0.0f;
        Wnt[j] = (_Float16)v;
    }
}

// ---------------------------------------------------------------------------
// Kernel 1: self embedding. One workgroup per batch sample b (class uniform).
// out(32x512) = A(32x128) @ Wself_f16[c](128x512) via 16x16x32 f16 WMMA.
// Bias is folded into the accumulator init (C-matrix splat: each D element of
// lane l has N = l%16). Invalid rows store to a dump zone via address select
// (v_cndmask) instead of exec-mask branches.
// ---------------------------------------------------------------------------
__global__ __launch_bounds__(256) void self_emb_kernel(
    const float* __restrict__ obs, const float* __restrict__ init_trajs,
    const _Float16* __restrict__ Wst, const float* __restrict__ b_self,
    const int* __restrict__ self_labels, float* __restrict__ xout,
    float* __restrict__ dump)
{
    __shared__ _Float16 A[32][DSELF_P];   // 8 KB
    const int b = blockIdx.x;
    const int t = threadIdx.x;
    const int c = self_labels[b];

    // Stage A (fp32 -> f16) with zero padding.
    for (int idx = t; idx < 32 * DSELF_P; idx += 256) {
        int row = idx >> 7, col = idx & 127;
        float v = 0.0f;
        if (row < KK && col < DSELF) {
            if (col < OBS * INs) v = obs[(size_t)b * OBS * INs + col];
            else v = init_trajs[((size_t)c * KK + row) * (PRED * INs) + (col - OBS * INs)];
        }
        A[row][col] = (_Float16)v;
    }
    __syncthreads();

    const int w = t >> 5, lane = t & 31;
    const int g = lane >> 4, lm = lane & 15;
    const int mt = w & 1;
    const int arow = mt * 16 + lm;
    float* dmp = dump + t;

    // Preload A fragments for the 4 K-steps (two ds_load_b128 each).
    Frag af[4];
#pragma unroll
    for (int kk = 0; kk < 4; ++kk) {
        const _Float16* p = &A[arow][kk * 32 + 8 * g];
        af[kk].q[0] = *(const v4u*)p;
        af[kk].q[1] = *(const v4u*)(p + 16);
    }

    const _Float16* Wc = Wst + (size_t)c * EE * DSELF_P;
    float* xb = xout + (size_t)b * KK * EE;
    const int nt0 = w >> 1;

#pragma unroll 2
    for (int i = 0; i < 8; ++i) {
        const int nt = nt0 + 4 * i;
        const int col = nt * 16 + lm;
        const _Float16* wp = Wc + (size_t)col * DSELF_P + 8 * g;

        Frag bf[4];
#pragma unroll
        for (int kk = 0; kk < 4; ++kk) {
            bf[kk].q[0] = *(const v4u*)(wp + kk * 32);
            bf[kk].q[1] = *(const v4u*)(wp + kk * 32 + 16);
        }

        const float bias = b_self[c * EE + col];
        v8f acc;
#pragma unroll
        for (int e8 = 0; e8 < 8; ++e8) acc[e8] = bias;   // bias as C-matrix

#pragma unroll
        for (int kk = 0; kk < 4; ++kk)
            acc = __builtin_amdgcn_wmma_f32_16x16x32_f16(
                false, af[kk].h, false, bf[kk].h, (short)0, acc, false, false);

#pragma unroll
        for (int e8 = 0; e8 < 8; ++e8) {
            int row = mt * 16 + e8 + 8 * g;
            float* dst = (row < KK) ? (xb + (size_t)row * EE + col) : dmp;
            *dst = acc[e8];
        }
    }
}

// ---------------------------------------------------------------------------
// Kernel 2: neighbor embedding. One workgroup per batch sample b.
// Per-row classes -> thread 0 builds a class-grouped row permutation (each
// class padded to a multiple of 16; pad slots map to an all-zero LDS row so
// the A-fragment gather is branch-free). D rows scatter back through rowmap;
// pad rows store to the dump zone via address select.
// ---------------------------------------------------------------------------
__global__ __launch_bounds__(256) void nei_emb_kernel(
    const float* __restrict__ neis, const _Float16* __restrict__ Wnt,
    const float* __restrict__ b_nei, const int* __restrict__ nei_labels,
    float* __restrict__ nout, float* __restrict__ dump)
{
    __shared__ _Float16 A[NN + 1][DNEI_P];   // row NN is all-zero pad row
    __shared__ int rowmap[192];
    __shared__ int tilecls[12];
    __shared__ int T_s;

    const int b = blockIdx.x;
    const int t = threadIdx.x;

    // Stage signed-reciprocal-transformed neighbors (fp32 -> f16), K padded,
    // plus one all-zero row at index NN for padding slots.
    for (int idx = t; idx < (NN + 1) * DNEI_P; idx += 256) {
        int n = idx >> 6, k = idx & 63;
        float v = 0.0f;
        if (n < NN && k < DNEI) {
            float x = neis[((size_t)b * NN + n) * DNEI + k];
            v = (x >= 0.0f) ? 1.0f / (x + 1e-4f) : 1.0f / (x - 1e-4f);
        }
        A[n][k] = (_Float16)v;
    }

    if (t == 0) {
        int cnt[CN], pos[CN], labs[NN];
        for (int c = 0; c < CN; ++c) cnt[c] = 0;
        const int* lab = nei_labels + (size_t)b * NN;
        for (int n = 0; n < NN; ++n) {
            int c = lab[n];
            c = c < 0 ? 0 : (c >= CN ? CN - 1 : c);
            labs[n] = c;
            cnt[c]++;
        }
        int s = 0, T = 0;
        for (int c = 0; c < CN; ++c) {
            pos[c] = s;
            int tc = (cnt[c] + 15) >> 4;        // padded tile count
            for (int i = 0; i < tc; ++i) tilecls[T + i] = c;
            s += tc * 16;
            T += tc;
        }
        for (int i = 0; i < s; ++i) rowmap[i] = NN;   // pad -> zero LDS row
        for (int n = 0; n < NN; ++n) rowmap[pos[labs[n]]++] = n;
        T_s = T;
    }
    __syncthreads();

    const int T = T_s;                        // 4..11 tiles
    const int w = t >> 5, lane = t & 31;
    const int g = lane >> 4, lm = lane & 15;
    float* ob = nout + (size_t)b * NN * EE;
    float* dmp = dump + t;

    // Incremental (tile, nt) walk over T*32 jobs, stride 8 waves: avoids the
    // per-iteration integer divide.
    int tile = w, nt = 0;
    while (tile >= T) { tile -= T; ++nt; }

    while (nt < 32) {
        const int c = tilecls[tile];
        const int col = nt * 16 + lm;

        // Issue global B-fragment loads first; LDS gather hides their latency.
        const _Float16* wp = Wnt + ((size_t)c * EE + col) * DNEI_P + 8 * g;
        Frag bf0, bf1;
        bf0.q[0] = *(const v4u*)wp;
        bf0.q[1] = *(const v4u*)(wp + 16);
        bf1.q[0] = *(const v4u*)(wp + 32);
        bf1.q[1] = *(const v4u*)(wp + 48);
        const float bias = b_nei[c * EE + col];

        // Branch-free A gather: pad slots point at the zero row.
        const int n = rowmap[tile * 16 + lm];
        const _Float16* p = &A[n][8 * g];
        Frag af0, af1;
        af0.q[0] = *(const v4u*)p;
        af0.q[1] = *(const v4u*)(p + 16);
        af1.q[0] = *(const v4u*)(p + 32);
        af1.q[1] = *(const v4u*)(p + 48);

        v8f acc;
#pragma unroll
        for (int e8 = 0; e8 < 8; ++e8) acc[e8] = bias;   // bias as C-matrix

        acc = __builtin_amdgcn_wmma_f32_16x16x32_f16(
            false, af0.h, false, bf0.h, (short)0, acc, false, false);
        acc = __builtin_amdgcn_wmma_f32_16x16x32_f16(
            false, af1.h, false, bf1.h, (short)0, acc, false, false);

#pragma unroll
        for (int e8 = 0; e8 < 8; ++e8) {
            int nrow = rowmap[tile * 16 + e8 + 8 * g];
            float* dst = (nrow < NN) ? (ob + (size_t)nrow * EE + col) : dmp;
            *dst = acc[e8];
        }

        tile += 8;
        while (tile >= T) { tile -= T; ++nt; }
    }
}

// ---------------------------------------------------------------------------
// Launch: prep (weights fp32->f16 transposed in d_ws, ~1.22MB + 1KB dump),
// then the two WMMA kernels. d_out = [x (B*K*E) | nei_feats (B*N*E)] fp32.
// ---------------------------------------------------------------------------
extern "C" void kernel_launch(void* const* d_in, const int* in_sizes, int n_in,
                              void* d_out, int out_size, void* d_ws, size_t ws_size,
                              hipStream_t stream) {
    const float* obs         = (const float*)d_in[0];
    const float* neis        = (const float*)d_in[1];
    const float* init_trajs  = (const float*)d_in[2];
    const float* W_self      = (const float*)d_in[3];
    const float* b_self      = (const float*)d_in[4];
    const float* W_nei       = (const float*)d_in[5];
    const float* b_nei       = (const float*)d_in[6];
    const int*   self_labels = (const int*)d_in[7];
    const int*   nei_labels  = (const int*)d_in[8];

    float* x_out  = (float*)d_out;                       // [B, K, E]
    float* nf_out = x_out + (size_t)BSZ * KK * EE;       // [B, N, E]

    _Float16* Wst = (_Float16*)d_ws;                     // [C][E][128]
    _Float16* Wnt = Wst + WST_ELEMS;                     // [C+1][E][64]
    float* dump   = (float*)((char*)d_ws + DUMP_OFF);    // 256-float sink

    const int prep_total = WST_ELEMS + WNT_ELEMS;
    prep_weights_kernel<<<(prep_total + 255) / 256, 256, 0, stream>>>(
        W_self, W_nei, Wst, Wnt);
    self_emb_kernel<<<BSZ, 256, 0, stream>>>(
        obs, init_trajs, Wst, b_self, self_labels, x_out, dump);
    nei_emb_kernel<<<BSZ, 256, 0, stream>>>(
        neis, Wnt, b_nei, nei_labels, nf_out, dump);
}
